// MultidomainEncoder_25177098289227
// MI455X (gfx1250) — compile-verified, matrix operand order checked by reading, not run on hardware
//
#include <hip/hip_runtime.h>

// ---------------------------------------------------------------------------
// MultidomainEncoder forward for MI455X (gfx1250, wave32, WMMA).
// Heavy math: v_wmma_f32_16x16x32_f16 (f32 accumulate).
// Weight tiles: GLOBAL_LOAD_ASYNC_TO_LDS_B128 (ASYNCcnt), double-buffered.
// ---------------------------------------------------------------------------

typedef __attribute__((ext_vector_type(16))) _Float16 v16h;
typedef __attribute__((ext_vector_type(8)))  float    v8f;

#define HD    512              // model width H
#define SEQ   1024             // sequence length (N == M)
#define BATCH 4
#define ROWS  (BATCH * SEQ)    // 4096 flattened rows

// ---- WMMA helper ----------------------------------------------------------
__device__ __forceinline__ v8f wmma32(v16h a, v16h b, v8f c) {
  // 8 args: (neg_a, A, neg_b, B, c_mod, C, reuse_a, reuse_b)
  return __builtin_amdgcn_wmma_f32_16x16x32_f16(false, a, false, b, (short)0, c,
                                                false, false);
}

// ---- CDNA5 async global->LDS helpers --------------------------------------
// Generic pointers to __shared__ carry the LDS byte offset in their low 32
// bits (LDS aperture truncation, ISA 10.2) — exactly what VDST expects.
__device__ __forceinline__ unsigned lds_off(const void* p) {
  return (unsigned)(size_t)p;
}
__device__ __forceinline__ void async_load_b128(unsigned ldsaddr, const void* gaddr) {
  asm volatile("global_load_async_to_lds_b128 %0, %1, off"
               :: "v"(ldsaddr), "v"((unsigned long long)(size_t)gaddr)
               : "memory");
}
__device__ __forceinline__ void wait_async0() {
  asm volatile("s_wait_asynccnt 0x0" ::: "memory");
}

// A-fragment (16x32 f16, MxK) — and, by symmetry, B-fragment built from a
// K-contiguous transposed matrix (row = N, col = K). K order used here (same for A and B):
// element j of lane L:  row = L&15, k = (L>>4)*16 + j   (contiguous in memory)
__device__ __forceinline__ v16h frag_row(const _Float16* __restrict__ base,
                                         int ld, int r0, int c0, int lane) {
  const _Float16* p = base + (size_t)(r0 + (lane & 15)) * ld
                           + (size_t)(c0 + ((lane >> 4) << 4));
  v16h v;
#pragma unroll
  for (int j = 0; j < 16; ++j) v[j] = p[j];
  return v;
}

// B-fragment (32x16 f16, KxN) from a row-major [K, ld] matrix (e.g. V):
// element j of lane L:  k = (L>>4)*16 + j, n = L&15   (stride-ld gather)
__device__ __forceinline__ v16h frag_col(const _Float16* __restrict__ base,
                                         int ld, int k0, int c0, int lane) {
  const _Float16* p = base + (size_t)(k0 + ((lane >> 4) << 4)) * ld
                           + (size_t)(c0 + (lane & 15));
  v16h v;
#pragma unroll
  for (int j = 0; j < 16; ++j) v[j] = p[(size_t)j * ld];
  return v;
}

// ---- generic f16 GEMM: C[M,N](f32) = A[M,K](f16) x W^T[N,K](f16) ----------
// Block = 8 waves, 128(M) x 64(N) tile. Per 32-wide K-step the 64x32 weight
// tile is staged into LDS by async DMA (double-buffered); each wave reuses
// one A fragment across 4 WMMA accumulators.
__global__ void __launch_bounds__(256)
gemm_tn(const _Float16* __restrict__ A, int lda,
        const _Float16* __restrict__ WT, int ldw,
        const float* __restrict__ bias,
        float* __restrict__ Cf, _Float16* __restrict__ Ch, int ldc,
        int K, int acc) {
  const int tid  = threadIdx.x;
  const int lane = tid & 31;
  const int wave = tid >> 5;
  const int n0 = blockIdx.x * 64;
  const int m0 = (blockIdx.y * 8 + wave) * 16;

  __shared__ _Float16 wbuf[2][64 * 32];   // [n][k] weight tiles, 2 x 4KB

  // one b128 per thread per stage: row r = tid>>2 (0..63), col c = (tid&3)*8
  const int sr = tid >> 2;
  const int scc = (tid & 3) * 8;
  auto stage = [&](int buf, int k0) {
    const _Float16* src = WT + (size_t)(n0 + sr) * ldw + (k0 + scc);
    async_load_b128(lds_off(&wbuf[buf][sr * 32 + scc]), src);
  };

  v8f c0 = {}, c1 = {}, c2 = {}, c3 = {};
  const int nsteps = K / 32;
  stage(0, 0);
  for (int s = 0; s < nsteps; ++s) {
    wait_async0();          // this wave's tile DMA complete
    __syncthreads();        // all waves' tile DMA complete (and prev readers done)
    if (s + 1 < nsteps) stage((s + 1) & 1, (s + 1) * 32);  // overlap next DMA
    const _Float16* wt = wbuf[s & 1];
    v16h a = frag_row(A, lda, m0, s * 32, lane);
    c0 = wmma32(a, frag_row(wt, 32, 0,  0, lane), c0);
    c1 = wmma32(a, frag_row(wt, 32, 16, 0, lane), c1);
    c2 = wmma32(a, frag_row(wt, 32, 32, 0, lane), c2);
    c3 = wmma32(a, frag_row(wt, 32, 48, 0, lane), c3);
  }
  // C/D layout: element i -> row m0 + i + 8*(lane>>4), col ntile + (lane&15)
  const int m = m0 + 8 * (lane >> 4);
  v8f* accv[4] = { &c0, &c1, &c2, &c3 };
#pragma unroll
  for (int t = 0; t < 4; ++t) {
    const int n = n0 + t * 16 + (lane & 15);
    const float bv = bias ? bias[n] : 0.0f;
#pragma unroll
    for (int i = 0; i < 8; ++i) {
      size_t idx = (size_t)(m + i) * ldc + n;
      float v = (*accv[t])[i] + bv;
      if (Cf) {
        if (acc) v += Cf[idx];
        Cf[idx] = v;
      }
      if (Ch) Ch[idx] = (_Float16)v;
    }
  }
}

// ---- streaming (flash-style) attention ------------------------------------
// One wave handles 16 queries for one (b, h); keys streamed in blocks of 32.
// Accumulates softmax(QK^T/sqrt(d)) V  into Out (f32, merged [b,s,h*d] layout).
template <int DH>
__global__ void __launch_bounds__(128)
flash_attn(const _Float16* __restrict__ Q, const _Float16* __restrict__ Km,
           const _Float16* __restrict__ V, float* __restrict__ Out) {
  constexpr int NH = HD / DH;   // heads
  constexpr int DT = DH / 16;   // d-tiles for PV
  constexpr int KF = DH / 32;   // k-steps for QK^T
  const int lane = threadIdx.x & 31;
  const int wave = threadIdx.x >> 5;
  const int qt = blockIdx.x * 4 + wave;            // query tile in [0, SEQ/16)
  const int b  = blockIdx.y / NH;
  const int h  = blockIdx.y % NH;
  const size_t base = ((size_t)b * SEQ) * HD + (size_t)h * DH;
  const _Float16* Qb = Q + base;
  const _Float16* Kb = Km + base;
  const _Float16* Vb = V + base;
  float* Ob = Out + base;
  const int m0 = qt * 16;
  const float scale = 1.0f / sqrtf((float)DH);

  __shared__ _Float16 lds[4][16 * 32];             // per-wave P staging
  _Float16* P = lds[wave];

  v8f O[DT];
#pragma unroll
  for (int t = 0; t < DT; ++t) O[t] = (v8f){};
  float m_i[8], l_i[8];
#pragma unroll
  for (int i = 0; i < 8; ++i) { m_i[i] = -1e30f; l_i[i] = 0.0f; }

  v16h qf[KF];
#pragma unroll
  for (int kk = 0; kk < KF; ++kk) qf[kk] = frag_row(Qb, HD, m0, kk * 32, lane);

  for (int n0 = 0; n0 < SEQ; n0 += 32) {
    // S = Q . K^T for 16 queries x 32 keys (two 16x16 score tiles)
    v8f s0 = {}, s1 = {};
#pragma unroll
    for (int kk = 0; kk < KF; ++kk) {
      s0 = wmma32(qf[kk], frag_row(Kb, HD, n0,      kk * 32, lane), s0);
      s1 = wmma32(qf[kk], frag_row(Kb, HD, n0 + 16, kk * 32, lane), s1);
    }
    // online softmax: per accumulator element i, row = i + 8*(lane>>4);
    // reductions stay inside each 16-lane half (xor masks 1..8).
    float ef[8];
#pragma unroll
    for (int i = 0; i < 8; ++i) {
      float a0 = s0[i] * scale;
      float a1 = s1[i] * scale;
      float rm = fmaxf(a0, a1);
#pragma unroll
      for (int o = 1; o < 16; o <<= 1) rm = fmaxf(rm, __shfl_xor(rm, o, 32));
      float mn = fmaxf(m_i[i], rm);
      float e  = __expf(m_i[i] - mn);
      float p0 = __expf(a0 - mn);
      float p1 = __expf(a1 - mn);
      float rs = p0 + p1;
#pragma unroll
      for (int o = 1; o < 16; o <<= 1) rs += __shfl_xor(rs, o, 32);
      l_i[i] = l_i[i] * e + rs;
      m_i[i] = mn;
      ef[i]  = e;
      s0[i] = p0;
      s1[i] = p1;
    }
#pragma unroll
    for (int t = 0; t < DT; ++t)
#pragma unroll
      for (int i = 0; i < 8; ++i) O[t][i] *= ef[i];

    // re-stripe P from C-layout to A-layout via LDS (f16), 16 rows x 32 keys
#pragma unroll
    for (int i = 0; i < 8; ++i) {
      int r = i + 8 * (lane >> 4);
      P[r * 32 + (lane & 15)]      = (_Float16)s0[i];
      P[r * 32 + 16 + (lane & 15)] = (_Float16)s1[i];
    }
    asm volatile("s_wait_dscnt 0x0" ::: "memory");
    v16h pa;
    {
      const _Float16* pp = P + (lane & 15) * 32 + ((lane >> 4) << 4);
#pragma unroll
      for (int j = 0; j < 16; ++j) pa[j] = pp[j];
    }
    // O += P . V  (K-dim = 32 keys)
#pragma unroll
    for (int t = 0; t < DT; ++t)
      O[t] = wmma32(pa, frag_col(Vb, HD, n0, t * 16, lane), O[t]);
    asm volatile("s_wait_dscnt 0x0" ::: "memory");
  }
  // epilogue: divide by l, accumulate into merged output
#pragma unroll
  for (int t = 0; t < DT; ++t)
#pragma unroll
    for (int i = 0; i < 8; ++i) {
      int r = m0 + i + 8 * (lane >> 4);
      Ob[(size_t)r * HD + t * 16 + (lane & 15)] += O[t][i] / l_i[i];
    }
}

// ---- small utility kernels ------------------------------------------------
__global__ void wconv_k(const float* __restrict__ W, _Float16* __restrict__ WT,
                        int K, int N) {  // W[K,N] f32 -> WT[N,K] f16
  int idx = blockIdx.x * blockDim.x + threadIdx.x;
  if (idx >= K * N) return;
  int k = idx / N, n = idx % N;
  WT[(size_t)n * K + k] = (_Float16)W[idx];
}

__global__ void cvt_k(const float* __restrict__ x, _Float16* __restrict__ y, int n) {
  int i = blockIdx.x * blockDim.x + threadIdx.x;
  if (i < n) y[i] = (_Float16)x[i];
}

__global__ void zero_k(float* __restrict__ x, int n) {
  int i = blockIdx.x * blockDim.x + threadIdx.x;
  if (i < n) x[i] = 0.0f;
}

// e = concat([cx, y[:, :, slice, :]], -1) @ in_W + in_b      (K = 3)
__global__ void proj3_k(const float* __restrict__ cx, const float* __restrict__ cy,
                        int slice, const float* __restrict__ W,
                        const float* __restrict__ bias,
                        float* __restrict__ of, _Float16* __restrict__ oh) {
  int row = blockIdx.y;
  int j = blockIdx.x * blockDim.x + threadIdx.x;
  float x0 = cx[row];
  float y0 = cy[(size_t)row * 8 + slice * 2 + 0];
  float y1 = cy[(size_t)row * 8 + slice * 2 + 1];
  float v = x0 * W[j] + y0 * W[HD + j] + y1 * W[2 * HD + j] + bias[j];
  size_t idx = (size_t)row * HD + j;
  if (of) of[idx] = v;
  if (oh) oh[idx] = (_Float16)v;
}

// out = x @ W + b   with K = 1
__global__ void proj1_k(const float* __restrict__ x, const float* __restrict__ W,
                        const float* __restrict__ bias,
                        float* __restrict__ of, _Float16* __restrict__ oh) {
  int row = blockIdx.y;
  int j = blockIdx.x * blockDim.x + threadIdx.x;
  float v = x[row] * W[j] + bias[j];
  size_t idx = (size_t)row * HD + j;
  if (of) of[idx] = v;
  if (oh) oh[idx] = (_Float16)v;
}

// layernorm(f + res) * g + beta, per row of 512; one wave per row
__global__ void __launch_bounds__(32)
ln_k(const float* __restrict__ f, const float* __restrict__ res,
     const float* __restrict__ g, const float* __restrict__ beta,
     float* __restrict__ of, _Float16* __restrict__ oh) {
  const int row = blockIdx.x;
  const int lane = threadIdx.x;
  float vals[16];
  float s = 0.0f;
#pragma unroll
  for (int i = 0; i < 16; ++i) {
    int c = lane + 32 * i;
    float v = f[(size_t)row * HD + c] + res[(size_t)row * HD + c];
    vals[i] = v;
    s += v;
  }
#pragma unroll
  for (int o = 1; o < 32; o <<= 1) s += __shfl_xor(s, o, 32);
  float mean = s * (1.0f / HD);
  float vs = 0.0f;
#pragma unroll
  for (int i = 0; i < 16; ++i) { float d = vals[i] - mean; vs += d * d; }
#pragma unroll
  for (int o = 1; o < 32; o <<= 1) vs += __shfl_xor(vs, o, 32);
  float inv = rsqrtf(vs * (1.0f / HD) + 1e-5f);
#pragma unroll
  for (int i = 0; i < 16; ++i) {
    int c = lane + 32 * i;
    float outv = (vals[i] - mean) * inv * g[c] + beta[c];
    of[(size_t)row * HD + c] = outv;
    if (oh) oh[(size_t)row * HD + c] = (_Float16)outv;
  }
}

// ---------------------------------------------------------------------------
extern "C" void kernel_launch(void* const* d_in, const int* in_sizes, int n_in,
                              void* d_out, int out_size, void* d_ws, size_t ws_size,
                              hipStream_t stream) {
  (void)in_sizes; (void)n_in; (void)out_size; (void)ws_size;
  const float* context_x = (const float*)d_in[0];
  const float* context_y = (const float*)d_in[1];
  const float* target_x  = (const float*)d_in[2];
  const float* in_W  = (const float*)d_in[3];
  const float* in_b  = (const float*)d_in[4];
  const float* cx_W  = (const float*)d_in[5];
  const float* cx_b  = (const float*)d_in[6];
  const float* tx_W  = (const float*)d_in[7];
  const float* tx_b  = (const float*)d_in[8];
  const float* sc_Wk = (const float*)d_in[9];
  const float* sc_Wv = (const float*)d_in[10];
  const float* sc_Wq = (const float*)d_in[11];
  const float* sc_Wf = (const float*)d_in[12];
  const float* sc_bf = (const float*)d_in[13];
  const float* sc_g  = (const float*)d_in[14];
  const float* sc_be = (const float*)d_in[15];
  const float* ca_Wk = (const float*)d_in[16];
  const float* ca_Wv = (const float*)d_in[17];
  const float* ca_Wq = (const float*)d_in[18];
  const float* ca_Wf = (const float*)d_in[19];
  const float* ca_bf = (const float*)d_in[20];
  const float* ca_g  = (const float*)d_in[21];
  const float* ca_be = (const float*)d_in[22];

  // ---- workspace layout ----
  char* ws = (char*)d_ws;
  size_t off = 0;
  auto alloc = [&](size_t bytes) -> void* {
    void* p = ws + off;
    off += (bytes + 255) & ~(size_t)255;
    return p;
  };
  const size_t NE = (size_t)ROWS * HD;
  float*    e_f  = (float*)alloc(NE * 4);
  _Float16* e_h  = (_Float16*)alloc(NE * 2);
  _Float16* e1_h = (_Float16*)alloc(NE * 2);
  _Float16* e2_h = (_Float16*)alloc(NE * 2);
  _Float16* e3_h = (_Float16*)alloc(NE * 2);
  _Float16* Ks_h = (_Float16*)alloc(NE * 2);
  _Float16* Vs_h = (_Float16*)alloc(NE * 2);
  _Float16* Qs_h = (_Float16*)alloc(NE * 2);
  _Float16* Kc_h = (_Float16*)alloc(NE * 2);
  _Float16* Vc_h = (_Float16*)alloc(NE * 2);
  _Float16* Qc_h = (_Float16*)alloc(NE * 2);
  float*    r_f  = (float*)alloc(NE * 4);
  _Float16* r_h  = (_Float16*)alloc(NE * 2);
  float*    f_f  = (float*)alloc(NE * 4);
  float*    q_f  = (float*)alloc(NE * 4);
  _Float16* q_h  = (_Float16*)alloc(NE * 2);
  _Float16* k_h  = (_Float16*)alloc(NE * 2);
  _Float16 *scWkT[2], *scWvT[2], *scWqT[2], *scWfT[2];
  _Float16 *caWkT[2], *caWvT[2], *caWqT[2], *caWfT[2];
  for (int i = 0; i < 2; ++i) {
    scWkT[i] = (_Float16*)alloc((size_t)HD * HD * 2);
    scWvT[i] = (_Float16*)alloc((size_t)HD * HD * 2);
    scWqT[i] = (_Float16*)alloc((size_t)HD * HD * 2);
    scWfT[i] = (_Float16*)alloc((size_t)2 * HD * HD * 2);
    caWkT[i] = (_Float16*)alloc((size_t)HD * HD * 2);
    caWvT[i] = (_Float16*)alloc((size_t)HD * HD * 2);
    caWqT[i] = (_Float16*)alloc((size_t)HD * HD * 2);
    caWfT[i] = (_Float16*)alloc((size_t)2 * HD * HD * 2);
  }

  // ---- weight convert + transpose (once per launch, deterministic) ----
  auto wconv = [&](const float* W, _Float16* WT, int K, int N) {
    int n = K * N;
    wconv_k<<<dim3((n + 255) / 256), dim3(256), 0, stream>>>(W, WT, K, N);
  };
  for (int i = 0; i < 2; ++i) {
    wconv(sc_Wk + (size_t)i * HD * HD, scWkT[i], HD, HD);
    wconv(sc_Wv + (size_t)i * HD * HD, scWvT[i], HD, HD);
    wconv(sc_Wq + (size_t)i * HD * HD, scWqT[i], HD, HD);
    wconv(sc_Wf + (size_t)i * 2 * HD * HD, scWfT[i], 2 * HD, HD);
    wconv(ca_Wk + (size_t)i * HD * HD, caWkT[i], HD, HD);
    wconv(ca_Wv + (size_t)i * HD * HD, caWvT[i], HD, HD);
    wconv(ca_Wq + (size_t)i * HD * HD, caWqT[i], HD, HD);
    wconv(ca_Wf + (size_t)i * 2 * HD * HD, caWfT[i], 2 * HD, HD);
  }

  // ---- input projections ----
  dim3 pg(HD / 128, ROWS);
  proj3_k<<<pg, 128, 0, stream>>>(context_x, context_y, 0, in_W, in_b, e_f, e_h);
  proj3_k<<<pg, 128, 0, stream>>>(context_x, context_y, 1, in_W, in_b, nullptr, e1_h);
  proj3_k<<<pg, 128, 0, stream>>>(context_x, context_y, 2, in_W, in_b, nullptr, e2_h);
  proj3_k<<<pg, 128, 0, stream>>>(context_x, context_y, 3, in_W, in_b, nullptr, e3_h);
  proj1_k<<<pg, 128, 0, stream>>>(target_x, tx_W, tx_b, q_f, q_h);
  proj1_k<<<pg, 128, 0, stream>>>(context_x, cx_W, cx_b, nullptr, k_h);

  // ---- launch helpers ----
  auto gemm = [&](const _Float16* A, const _Float16* WT, int ldw,
                  const float* bias, float* Cf, _Float16* Ch, int K, int acc) {
    dim3 g(HD / 64, ROWS / 128);
    gemm_tn<<<g, 256, 0, stream>>>(A, HD, WT, ldw, bias, Cf, Ch, HD, K, acc);
  };
  auto zero = [&](float* p) {
    zero_k<<<dim3((int)(NE / 256)), 256, 0, stream>>>(p, (int)NE);
  };
  auto cvt = [&](const float* src, _Float16* dst) {
    cvt_k<<<dim3((int)(NE / 256)), 256, 0, stream>>>(src, dst, (int)NE);
  };
  auto flash_sc = [&](const _Float16* Qp, const _Float16* Kp, const _Float16* Vp) {
    flash_attn<256><<<dim3(SEQ / 64, BATCH * 2), 128, 0, stream>>>(Qp, Kp, Vp, r_f);
  };
  auto flash_ca = [&](const _Float16* Qp, const _Float16* Kp, const _Float16* Vp) {
    flash_attn<128><<<dim3(SEQ / 64, BATCH * 4), 128, 0, stream>>>(Qp, Kp, Vp, r_f);
  };

  // ---- one _sc_attn pass (x = e; x2 = nullptr means x2 == x) ----
  auto sc_pass = [&](int i, const _Float16* x2) {
    gemm(e_h, scWkT[i], HD, nullptr, nullptr, Ks_h, HD, 0);
    gemm(e_h, scWvT[i], HD, nullptr, nullptr, Vs_h, HD, 0);
    gemm(e_h, scWqT[i], HD, nullptr, nullptr, Qs_h, HD, 0);
    const _Float16 *Kc = Ks_h, *Vc = Vs_h, *Qc = Qs_h;
    if (x2) {
      gemm(x2, scWkT[i], HD, nullptr, nullptr, Kc_h, HD, 0);
      gemm(x2, scWvT[i], HD, nullptr, nullptr, Vc_h, HD, 0);
      gemm(x2, scWqT[i], HD, nullptr, nullptr, Qc_h, HD, 0);
      Kc = Kc_h; Vc = Vc_h; Qc = Qc_h;
    }
    zero(r_f);
    flash_sc(Qs_h, Ks_h, Vs_h);   // sdpa(sQ,sK,sV)
    flash_sc(Qc,   Kc,   Vc);     // sdpa(cQ,cK,cV)
    flash_sc(Qc,   Ks_h, Vs_h);   // sdpa(cQ,sK,sV)
    flash_sc(Qs_h, Kc,   Vc);     // sdpa(sQ,cK,cV)
    cvt(r_f, r_h);
    // concat([residual, r]) @ Wf + bf  ==  e @ Wf[:H] + r @ Wf[H:] + bf
    gemm(e_h, scWfT[i],      2 * HD, sc_bf + (size_t)i * HD, f_f, nullptr, HD, 0);
    gemm(r_h, scWfT[i] + HD, 2 * HD, nullptr,                f_f, nullptr, HD, 1);
    ln_k<<<ROWS, 32, 0, stream>>>(f_f, e_f, sc_g + (size_t)i * HD,
                                  sc_be + (size_t)i * HD, e_f, e_h);
  };

  // ---- one cross-attention pass ----
  auto ca_pass = [&](int i) {
    gemm(k_h, caWkT[i], HD, nullptr, nullptr, Ks_h, HD, 0);   // K from context proj
    gemm(e_h, caWvT[i], HD, nullptr, nullptr, Vs_h, HD, 0);   // V from encoder out
    gemm(q_h, caWqT[i], HD, nullptr, nullptr, Qs_h, HD, 0);   // Q from target proj
    zero(r_f);
    flash_ca(Qs_h, Ks_h, Vs_h);
    cvt(r_f, r_h);
    gemm(q_h, caWfT[i],      2 * HD, ca_bf + (size_t)i * HD, f_f, nullptr, HD, 0);
    gemm(r_h, caWfT[i] + HD, 2 * HD, nullptr,                f_f, nullptr, HD, 1);
    float* dst = (i == 1) ? (float*)d_out : q_f;
    ln_k<<<ROWS, 32, 0, stream>>>(f_f, q_f, ca_g + (size_t)i * HD,
                                  ca_be + (size_t)i * HD, dst, q_h);
  };

  // ---- full forward ----
  sc_pass(0, nullptr);
  sc_pass(1, nullptr);
  const _Float16* xs[3] = { e1_h, e2_h, e3_h };
  for (int s = 0; s < 3; ++s) {
    sc_pass(0, xs[s]);
    sc_pass(1, xs[s]);
  }
  ca_pass(0);
  ca_pass(1);
}
